// GateModLinear_60224031425144
// MI455X (gfx1250) — matrix operand order repaired and checked by reading, not compiled, not run on hardware
//
#include <hip/hip_runtime.h>
#include <hip/hip_bf16.h>

// Problem constants (from reference)
#define B_    4096
#define M_    8
#define DI_   2048
#define DO_   2048
#define K_    (M_ * DI_)      // 16384 : folded expert*feature contraction dim
#define LN_EPSF 1e-5f

// GEMM tiling
#define TILE_B 128
#define TILE_N 128
#define TILE_K 32
#define KT_    (K_ / TILE_K)  // 512
#define LDSTR  40             // bf16 elems per LDS row (32 data + 8 pad) -> 80B, bank-conflict-free

typedef __attribute__((ext_vector_type(16))) __bf16 v16bf;
typedef __attribute__((ext_vector_type(8)))  __bf16 bf16x8;   // 16 bytes
typedef __attribute__((ext_vector_type(4)))  __bf16 bf16x4;   // 8 bytes
typedef __attribute__((ext_vector_type(8)))  float  v8f;
typedef int v4i_ __attribute__((vector_size(16)));            // b128 payload type for async-LDS builtin

union FragAB { v16bf v; uint4 q[2]; };

// CDNA5 async global->LDS path (ASYNCcnt-tracked), guarded so a missing builtin
// degrades to the proven sync staging instead of breaking the compile.
#if __has_builtin(__builtin_amdgcn_global_load_async_to_lds_b128) && \
    __has_builtin(__builtin_amdgcn_s_wait_asynccnt)
#define HAVE_ASYNC_LDS 1
#else
#define HAVE_ASYNC_LDS 0
#endif

// ---------------------------------------------------------------------------
// Kernel 1: Ws[m][i][j] (f32) -> Wb[i][m*DI+j] (bf16).  Coalesced along j both ways.
// ---------------------------------------------------------------------------
__global__ __launch_bounds__(256)
void convert_W(const float* __restrict__ Ws, __bf16* __restrict__ Wb) {
    int idx = blockIdx.x * 256 + threadIdx.x;        // one float4 per thread
    size_t e = (size_t)idx * 4;                      // element index in Ws
    int m   = (int)(e >> 22);                        // / (DO_*DI_) = 2^22
    int rem = (int)(e & ((1u << 22) - 1u));
    int i   = rem >> 11;                             // / DI_
    int j   = rem & (DI_ - 1);
    float4 w = *(const float4*)(Ws + e);
    bf16x4 o;
    o[0] = (__bf16)w.x; o[1] = (__bf16)w.y; o[2] = (__bf16)w.z; o[3] = (__bf16)w.w;
    *(bf16x4*)(Wb + (size_t)i * K_ + (size_t)m * DI_ + j) = o;
}

// ---------------------------------------------------------------------------
// Kernel 2: C[b,i] = sum_k A'[b,k] * Wb[i,k],  A'[b,k] = pW[b, k>>11] * x[b, k&2047]
// 128x128 WG tile, 8 wave32s (2x4), each wave 64x32 = 4x2 WMMA 16x16 tiles.
// Double-buffered LDS; W' tiles DMA'd with global_load_async_to_lds_b128,
// A' tiles built in-register (pW*x -> bf16) and ds_store'd.
// ---------------------------------------------------------------------------
__global__ __launch_bounds__(256)
void gemm_moe(const float* __restrict__ x, const float* __restrict__ pW,
              const __bf16* __restrict__ Wb, float* __restrict__ Zb) {
    __shared__ __bf16 lA[2][TILE_B * LDSTR];
    __shared__ __bf16 lB[2][TILE_N * LDSTR];

    const int tid     = threadIdx.x;
    const int lane    = tid & 31;
    const int wave    = tid >> 5;
    const int waveRow = wave >> 2;     // 0..1  (64 rows each)
    const int waveCol = wave & 3;      // 0..3  (32 cols each)
    const int b0 = blockIdx.y * TILE_B;
    const int n0 = blockIdx.x * TILE_N;

    const int lhalf = lane >> 4;       // 0 / 1
    const int lm    = lane & 15;

    v8f acc[4][2];
#pragma unroll
    for (int ta = 0; ta < 4; ++ta)
#pragma unroll
        for (int tb = 0; tb < 2; ++tb)
#pragma unroll
            for (int v = 0; v < 8; ++v) acc[ta][tb][v] = 0.0f;

    // stage one K-step
    auto stage = [&](int buf, int kt) {
        const int k0 = kt * TILE_K;
#pragma unroll
        for (int it = 0; it < 2; ++it) {
            int task = it * 256 + tid;           // 512 tasks: 128 rows x 4 chunks of 8 bf16
            int r    = task >> 2;
            int c8   = (task & 3) * 8;
            int k    = k0 + c8;
            // ---- W' tile: row = output col n0+r (K-contiguous, bf16, L2-resident)
            const __bf16* wp = Wb + (size_t)(n0 + r) * K_ + k;
#if HAVE_ASYNC_LDS
            __builtin_amdgcn_global_load_async_to_lds_b128(
                (__attribute__((address_space(1))) v4i_*)wp,
                (__attribute__((address_space(3))) v4i_*)&lB[buf][r * LDSTR + c8],
                0, 0);
#else
            *(bf16x8*)&lB[buf][r * LDSTR + c8] = *(const bf16x8*)wp;
#endif
            __builtin_prefetch(wp + 8 * TILE_K, 0, 1);
            // ---- A' tile: row = batch b0+r, generated on the fly
            int   m  = k >> 11;                  // expert id (constant across the 8-elem chunk)
            int   j  = k & (DI_ - 1);
            int   b  = b0 + r;
            float pw = pW[b * M_ + m];
            float4 x0 = *(const float4*)(x + (size_t)b * DI_ + j);
            float4 x1 = *(const float4*)(x + (size_t)b * DI_ + j + 4);
            float  xs[8] = {x0.x, x0.y, x0.z, x0.w, x1.x, x1.y, x1.z, x1.w};
            bf16x8 a8;
#pragma unroll
            for (int q = 0; q < 8; ++q) a8[q] = (__bf16)(xs[q] * pw);
            *(bf16x8*)&lA[buf][r * LDSTR + c8] = a8;
        }
    };

    stage(0, 0);
#if HAVE_ASYNC_LDS
    __builtin_amdgcn_s_wait_asynccnt(0);
#endif
    __syncthreads();

    int buf = 0;
    for (int kt = 0; kt < KT_; ++kt) {
        if (kt + 1 < KT_) stage(buf ^ 1, kt + 1);   // async copies fly during compute

        // --- B fragments (32x16): lane<16 -> N=lm, K 0..15 ; lane>=16 -> N=lm, K 16..31
        FragAB fb[2];
#pragma unroll
        for (int tb = 0; tb < 2; ++tb) {
            const __bf16* p = &lB[buf][(waveCol * 32 + tb * 16 + lm) * LDSTR + lhalf * 16];
            fb[tb].q[0] = *(const uint4*)p;
            fb[tb].q[1] = *(const uint4*)(p + 8);
        }
        // --- A fragments (16x32): lane<16 -> M=lm, K 0..15 ; lane>=16 -> M=lm, K 16..31
        FragAB fa[4];
#pragma unroll
        for (int ta = 0; ta < 4; ++ta) {
            const __bf16* p = &lA[buf][(waveRow * 64 + ta * 16 + lm) * LDSTR + lhalf * 16];
            fa[ta].q[0] = *(const uint4*)p;
            fa[ta].q[1] = *(const uint4*)(p + 8);
        }
        // --- 8 WMMAs per wave per K-step
#pragma unroll
        for (int ta = 0; ta < 4; ++ta)
#pragma unroll
            for (int tb = 0; tb < 2; ++tb)
                acc[ta][tb] = __builtin_amdgcn_wmma_f32_16x16x32_bf16(
                    false, fa[ta].v, false, fb[tb].v,
                    (short)0, acc[ta][tb], false, false);

#if HAVE_ASYNC_LDS
        __builtin_amdgcn_s_wait_asynccnt(0);        // next buffer's DMA done
#endif
        __syncthreads();
        buf ^= 1;
    }

    // Epilogue: C layout (16x16 f32): VGPR v -> M = v + 8*lhalf, N = lm
#pragma unroll
    for (int ta = 0; ta < 4; ++ta)
#pragma unroll
        for (int tb = 0; tb < 2; ++tb) {
            int i = n0 + waveCol * 32 + tb * 16 + lm;
#pragma unroll
            for (int v = 0; v < 8; ++v) {
                int b = b0 + waveRow * 64 + ta * 16 + lhalf * 8 + v;
                Zb[(size_t)b * DO_ + i] = acc[ta][tb][v];
            }
        }
}

// ---------------------------------------------------------------------------
// Kernel 3: z = gW*acc + gb*(pb . bs)  ->  LayerNorm(row) -> ELU
// ---------------------------------------------------------------------------
__global__ __launch_bounds__(256)
void ln_elu(const float* __restrict__ Zb, const float* __restrict__ pb,
            const float* __restrict__ bs, const float* __restrict__ gW,
            const float* __restrict__ gb, float* __restrict__ out) {
    __shared__ float zrow[DO_];
    __shared__ float red0[256];
    __shared__ float red1[256];

    const int b   = blockIdx.x;
    const int tid = threadIdx.x;

    float pbr[M_];
#pragma unroll
    for (int m = 0; m < M_; ++m) pbr[m] = pb[b * M_ + m];

    float s = 0.f, ss = 0.f;
#pragma unroll
    for (int c = 0; c < DO_ / 256; ++c) {
        int i = c * 256 + tid;
        float bias = 0.f;
#pragma unroll
        for (int m = 0; m < M_; ++m) bias += pbr[m] * bs[m * DO_ + i];
        size_t off = (size_t)b * DO_ + i;
        float z = gW[off] * Zb[off] + gb[off] * bias;
        zrow[i] = z;
        s  += z;
        ss += z * z;
    }
    red0[tid] = s; red1[tid] = ss;
    __syncthreads();
    for (int o = 128; o > 0; o >>= 1) {
        if (tid < o) { red0[tid] += red0[tid + o]; red1[tid] += red1[tid + o]; }
        __syncthreads();
    }
    const float mu  = red0[0] * (1.0f / DO_);
    const float var = red1[0] * (1.0f / DO_) - mu * mu;
    const float rs  = rsqrtf(var + LN_EPSF);
#pragma unroll
    for (int c = 0; c < DO_ / 256; ++c) {
        int i = c * 256 + tid;
        float y = (zrow[i] - mu) * rs;
        out[(size_t)b * DO_ + i] = (y > 0.f) ? y : expm1f(y);
    }
}

// ---------------------------------------------------------------------------
extern "C" void kernel_launch(void* const* d_in, const int* in_sizes, int n_in,
                              void* d_out, int out_size, void* d_ws, size_t ws_size,
                              hipStream_t stream) {
    const float* x  = (const float*)d_in[0];
    const float* Ws = (const float*)d_in[1];
    const float* bs = (const float*)d_in[2];
    const float* pW = (const float*)d_in[3];
    const float* pb = (const float*)d_in[4];
    const float* gW = (const float*)d_in[5];
    const float* gb = (const float*)d_in[6];
    float* out = (float*)d_out;

    // scratch: [ W' bf16  DO_*K_*2 = 64MB | Z f32  B_*DO_*4 = 32MB ]
    __bf16* Wb = (__bf16*)d_ws;
    float*  Zb = (float*)((char*)d_ws + (size_t)DO_ * K_ * 2);

    convert_W<<<(M_ * DO_ * DI_ / 4) / 256, 256, 0, stream>>>(Ws, Wb);

    dim3 g(DO_ / TILE_N, B_ / TILE_B);   // 16 x 32 workgroups
    gemm_moe<<<g, 256, 0, stream>>>(x, pW, Wb, Zb);

    ln_elu<<<B_, 256, 0, stream>>>(Zb, pb, bs, gW, gb, out);
}
